// SuperPointONNXWithKeypoints_49941879718101
// MI455X (gfx1250) — matrix operand order repaired and checked by reading, not compiled, not run on hardware
//
#include <hip/hip_runtime.h>
#include <hip/hip_fp16.h>
#include <stdint.h>

// ---------------------------------------------------------------------------
// SuperPoint forward for MI455X (gfx1250).
// f16 WMMA implicit-GEMM convs, NHWC f16 activations (inter-layer tensors fit
// in the 192MB L2), fragment-ready packed weights. Per wave, per K-chunk:
//   1x global_load_b128x2 (A) + 2x ds_load_b128x2 (B) + 2x v_wmma (16x16x32).
// Wave tile: 16 cout x 32 px; block tile: 64 cout x 64 px.
// ---------------------------------------------------------------------------

typedef __attribute__((ext_vector_type(16))) _Float16 v16h;
typedef __attribute__((ext_vector_type(8)))  float    v8f;

#define LDSW 68          // 64-pixel tile + 2 halo + 2 pad columns
#define TN   64          // pixels (one row segment) per block
#define CPAD 16          // LDS channel-stride pad (keeps 32B alignment, spreads banks)
#define KMAX 1024
#define MB   64          // couts per block

static __device__ __forceinline__ uint32_t pack_h2(_Float16 lo, _Float16 hi) {
  union { _Float16 h[2]; uint32_t u; } x;
  x.h[0] = lo; x.h[1] = hi;
  return x.u;
}

union Frag16 { v16h v; uint32_t u[8]; _Float16 h[16]; };

// ---------------------------------------------------------------------------
// Weight pack: fp32 OIHW -> fragment-ready f16 A tiles.
// Spatial-major K: k = r*CIN + c  (r = kh*ks+kw).
// apk[((q*coutPad + m)*2 + half)*16 + j] = W[m][k], with
//   p = j/2, hw = j&1, k = q*32 + half*16 + 2p + hw
// so each lane's whole A fragment is 16 contiguous halfwords (32B).
// ---------------------------------------------------------------------------
__global__ void pack_weights_kernel(const float* __restrict__ w, _Float16* __restrict__ dst,
                                    int cout, int coutPad, int cin, int ks)
{
  const int kt = cin * ks * ks;
  const int kp = (kt + 31) & ~31;
  const int total = kp * coutPad;        // == chunks*coutPad*32
  for (int t = blockIdx.x * blockDim.x + threadIdx.x; t < total;
       t += gridDim.x * blockDim.x) {
    const int j    = t & 15;
    const int half = (t >> 4) & 1;
    const int m    = (t >> 5) % coutPad;
    const int q    = (t >> 5) / coutPad;
    const int p    = j >> 1;
    const int hw   = j & 1;
    const int kb   = half * 16 + p * 2;
    const int k    = q * 32 + kb + hw;   // spatial-major K index
    float v = 0.0f;
    if (m < cout && k < kt) {
      const int r = k / cin;             // filter tap
      const int c = k - r * cin;         // input channel
      v = w[(size_t)m * kt + (size_t)c * ks * ks + r];   // OIHW
    }
    dst[t] = (_Float16)v;
  }
}

// ---------------------------------------------------------------------------
// Implicit-GEMM conv (KSxKS, pad=(KS==3), stride 1), fused bias+ReLU, NHWC.
// ---------------------------------------------------------------------------
template<int CIN, int KS, bool IN_F32, bool RELU, bool OUT_F32>
__global__ __launch_bounds__(256)
void conv_wmma_kernel(const void* __restrict__ inp,
                      const _Float16* __restrict__ apk,
                      const float* __restrict__ bias,
                      void* __restrict__ outp,
                      int H, int W, int coutReal, int coutBlocks)
{
  constexpr int KT     = CIN * KS * KS;
  constexpr int KP     = (KT + 31) & ~31;
  constexpr int CHUNKS = KP / 32;
  constexpr int CST    = CIN + CPAD;     // LDS channel stride (halfwords)
  const int coutPad = coutBlocks * MB;

  __shared__ _Float16 lin[KS * LDSW * CST] __attribute__((aligned(32)));

  const int b  = blockIdx.z / coutBlocks;
  const int cb = blockIdx.z % coutBlocks;
  const int y  = blockIdx.y;
  const int x0 = blockIdx.x * TN;

  // ---- stage input rows (NHWC f16, zero-padded halo) into LDS -------------
  const int pad   = (KS == 3) ? 1 : 0;
  const int usedW = TN + 2 * pad;
  if (CIN >= 32 && !IN_F32) {
    // vector staging: 8 channels (16B) per thread per step
    constexpr int C8 = CIN / 8;
    for (int t = threadIdx.x; t < KS * LDSW * C8; t += 256) {
      const int c8  = (t % C8) * 8;
      const int u   = t / C8;
      const int col = u % LDSW;
      const int row = u / LDSW;
      const int gx  = x0 + col - pad;
      const int gy  = y + row - pad;
      uint32_t d0 = 0, d1 = 0, d2 = 0, d3 = 0;
      if (col < usedW && gx >= 0 && gx < W && gy >= 0 && gy < H) {
        const uint32_t* src = (const uint32_t*)((const _Float16*)inp +
            ((size_t)(b * H + gy) * W + gx) * CIN + c8);
        d0 = src[0]; d1 = src[1]; d2 = src[2]; d3 = src[3];
      }
      uint32_t* dst = (uint32_t*)&lin[(size_t)(row * LDSW + col) * CST + c8];
      dst[0] = d0; dst[1] = d1; dst[2] = d2; dst[3] = d3;
    }
  } else {
    for (int t = threadIdx.x; t < KS * LDSW * CIN; t += 256) {
      const int c   = t % CIN;
      const int u   = t / CIN;
      const int col = u % LDSW;
      const int row = u / LDSW;
      const int gx  = x0 + col - pad;
      const int gy  = y + row - pad;
      float v = 0.0f;
      if (col < usedW && gx >= 0 && gx < W && gy >= 0 && gy < H) {
        const size_t gi = ((size_t)(b * H + gy) * W + gx) * CIN + c;
        v = IN_F32 ? ((const float*)inp)[gi] : (float)((const _Float16*)inp)[gi];
      }
      lin[(size_t)(row * LDSW + col) * CST + c] = (_Float16)v;
    }
  }
  __syncthreads();

  const int wave  = threadIdx.x >> 5;
  const int lane  = threadIdx.x & 31;
  const int half  = lane >> 4;
  const int nlane = lane & 15;
  const int mBase = cb * MB + (wave >> 1) * 16;   // 4 M-tiles per block
  const int px    = (wave & 1) * 32;              // 2 N-tiles (32 px) per block

  // per-lane A fragment pointer (16 contiguous halfwords per chunk)
  const _Float16* aptr = apk + ((size_t)(mBase + nlane) * 2 + half) * 16;
  const size_t aStep = (size_t)coutPad * 32;

  v8f acc0 = {}, acc1 = {};

#pragma unroll 4
  for (int q = 0; q < CHUNKS; ++q) {
    if (q + 1 < CHUNKS)   // gfx1250: lowers to global_prefetch_b8
      __builtin_prefetch(aptr + (q + 1) * aStep, 0, 1);

    Frag16 A, B0, B1;
    A.v = *(const v16h*)(aptr + q * aStep);

    if (CIN >= 32) {
      // chunk lies in one filter tap: r = q / (CIN/32)
      constexpr int CINC = CIN / 32;
      const int rr = q / CINC;
      const int cw = (q - rr * CINC) * 32;
      const int dy = (KS == 3) ? (rr / 3) : 0;
      const int dx = (KS == 3) ? (rr - 3 * (rr / 3)) : 0;
      const size_t rowOff = (size_t)(dy * LDSW + px + nlane + dx) * CST
                            + cw + half * 16;
      B0.v = *(const v16h*)&lin[rowOff];                      // px .. px+15
      B1.v = *(const v16h*)&lin[rowOff + (size_t)16 * CST];   // px+16 .. px+31
    } else {
      // layer-1a fallback (KT=9): scalar gather, k = r (CIN==1)
#pragma unroll
      for (int p = 0; p < 8; ++p) {
        const int kB = q * 32 + half * 16 + p * 2;
        _Float16 e0a = (_Float16)0.0f, e1a = (_Float16)0.0f;
        _Float16 e0b = (_Float16)0.0f, e1b = (_Float16)0.0f;
        if (kB < KT) {
          const int dy = kB / 3, dx = kB - 3 * dy;
          const size_t o = (size_t)(dy * LDSW + px + nlane + dx) * CST;
          e0a = lin[o];
          e0b = lin[o + (size_t)16 * CST];
        }
        if (kB + 1 < KT) {
          const int k1 = kB + 1;
          const int dy = k1 / 3, dx = k1 - 3 * dy;
          const size_t o = (size_t)(dy * LDSW + px + nlane + dx) * CST;
          e1a = lin[o];
          e1b = lin[o + (size_t)16 * CST];
        }
        B0.u[p] = pack_h2(e0a, e1a);
        B1.u[p] = pack_h2(e0b, e1b);
      }
    }
    acc0 = __builtin_amdgcn_wmma_f32_16x16x32_f16(false, A.v, false, B0.v,
                                                  (short)0, acc0, false, false);
    acc1 = __builtin_amdgcn_wmma_f32_16x16x32_f16(false, A.v, false, B1.v,
                                                  (short)0, acc1, false, false);
  }

  // ---- D layout: VGPR r -> M = r + 8*half, N = lane&15; NHWC store --------
  const int mo0 = mBase + half * 8;
#pragma unroll
  for (int sub = 0; sub < 2; ++sub) {
    const v8f acc = sub ? acc1 : acc0;
    const int xo = x0 + px + sub * 16 + nlane;
    float vr[8];
#pragma unroll
    for (int r = 0; r < 8; ++r) {
      const int mo = mo0 + r;
      float v = acc[r] + (mo < coutReal ? bias[mo] : 0.0f);
      if (RELU) v = fmaxf(v, 0.0f);
      vr[r] = v;
    }
    const size_t pixBase = ((size_t)(b * H + y) * W + xo) * coutReal;
    if (!OUT_F32 && (mo0 + 7 < coutReal) && ((coutReal & 7) == 0)) {
      // 8 contiguous f16 channels -> one 16B store
      uint32_t st[4];
#pragma unroll
      for (int i = 0; i < 4; ++i)
        st[i] = pack_h2((_Float16)vr[2 * i], (_Float16)vr[2 * i + 1]);
      uint32_t* dst = (uint32_t*)((_Float16*)outp + pixBase + mo0);
      dst[0] = st[0]; dst[1] = st[1]; dst[2] = st[2]; dst[3] = st[3];
    } else {
#pragma unroll
      for (int r = 0; r < 8; ++r) {
        const int mo = mo0 + r;
        if (mo < coutReal) {
          if (OUT_F32) ((float*)outp)[pixBase + mo] = vr[r];
          else         ((_Float16*)outp)[pixBase + mo] = (_Float16)vr[r];
        }
      }
    }
  }
}

// ---------------------------------------------------------------------------
// 2x2 max pool, f16 NHWC
// ---------------------------------------------------------------------------
__global__ void maxpool_f16_kernel(const _Float16* __restrict__ in, _Float16* __restrict__ out,
                                   int C, int Hi, int Wi)
{
  const int Ho = Hi >> 1, Wo = Wi >> 1;
  const size_t total = (size_t)2 * Ho * Wo * C;
  for (size_t t = (size_t)blockIdx.x * blockDim.x + threadIdx.x; t < total;
       t += (size_t)gridDim.x * blockDim.x) {
    const int c = (int)(t % C);
    size_t p = t / C;
    const int xo = (int)(p % Wo); p /= Wo;
    const int yo = (int)(p % Ho);
    const int bb = (int)(p / Ho);
    const size_t i00 = ((size_t)(bb * Hi + yo * 2) * Wi + xo * 2) * C + c;
    const float a = (float)in[i00];
    const float b = (float)in[i00 + C];
    const float d = (float)in[i00 + (size_t)Wi * C];
    const float e = (float)in[i00 + (size_t)Wi * C + C];
    out[t] = (_Float16)fmaxf(fmaxf(a, b), fmaxf(d, e));
  }
}

// ---------------------------------------------------------------------------
// Per-pixel L2 normalize of NHWC descriptor map d[B][64][64][256]
// ---------------------------------------------------------------------------
__global__ void normalize_d_kernel(float* __restrict__ d)
{
  const int t = blockIdx.x * blockDim.x + threadIdx.x;
  if (t >= 2 * 64 * 64) return;
  float* p = d + (size_t)t * 256;
  float ss = 0.0f;
  for (int c = 0; c < 256; ++c) { const float v = p[c]; ss += v * v; }
  const float inv = 1.0f / fmaxf(sqrtf(ss), 1e-12f);
  for (int c = 0; c < 256; ++c) p[c] *= inv;
}

// ---------------------------------------------------------------------------
// Softmax over 65 channels (NHWC), drop dustbin, depth-to-space 8x8, border.
// p[B][64][64][65] -> s[B][512][512]
// ---------------------------------------------------------------------------
__global__ void softmax_d2s_kernel(const float* __restrict__ p, float* __restrict__ s)
{
  const int t = blockIdx.x * blockDim.x + threadIdx.x;
  if (t >= 2 * 64 * 64) return;
  const int b = t >> 12, cell = t & 4095;
  const int yc = cell >> 6, xc = cell & 63;
  const float* pc = p + (size_t)t * 65;
  float mx = -1e30f;
  for (int c = 0; c < 65; ++c) mx = fmaxf(mx, pc[c]);
  float sum = 0.0f;
  for (int c = 0; c < 65; ++c) sum += __expf(pc[c] - mx);
  const float inv = 1.0f / sum;
  for (int c = 0; c < 64; ++c) {
    const int r = c >> 3, ccol = c & 7;
    const int yy = yc * 8 + r, xx = xc * 8 + ccol;
    float v = __expf(pc[c] - mx) * inv;
    if (yy < 4 || yy >= 508 || xx < 4 || xx >= 508) v = -1.0f;
    s[((size_t)b * 512 + yy) * 512 + xx] = v;
  }
}

// ---------------------------------------------------------------------------
// Top-K=1024 per batch: threshold binary search + compaction + bitonic sort.
// ---------------------------------------------------------------------------
__global__ __launch_bounds__(1024)
void topk_kernel(const float* __restrict__ s, float* __restrict__ kpts_out,
                 float* __restrict__ scores_out, float* __restrict__ kinfo)
{
  const int b = blockIdx.x;
  const float* sb = s + (size_t)b * 512 * 512;
  const int tid = threadIdx.x;
  const int N = 512 * 512;

  __shared__ int cnt;
  __shared__ float skey[KMAX];
  __shared__ int   sidx[KMAX];

  float lo = -2.0f, hi = 2.0f;
  for (int it = 0; it < 24; ++it) {
    const float mid = 0.5f * (lo + hi);
    if (tid == 0) cnt = 0;
    __syncthreads();
    int c = 0;
    for (int i = tid; i < N; i += 1024) c += (sb[i] > mid) ? 1 : 0;
    atomicAdd(&cnt, c);
    __syncthreads();
    const int total = cnt;
    __syncthreads();
    if (total > KMAX) lo = mid; else hi = mid;
  }
  const float thr = hi;

  if (tid == 0) cnt = 0;
  skey[tid] = -2.0f; sidx[tid] = 0;
  __syncthreads();
  for (int i = tid; i < N; i += 1024) {
    const float v = sb[i];
    if (v > thr) {
      const int pos = atomicAdd(&cnt, 1);
      if (pos < KMAX) { skey[pos] = v; sidx[pos] = i; }
    }
  }
  __syncthreads();

  for (int size = 2; size <= KMAX; size <<= 1) {
    for (int stride = size >> 1; stride > 0; stride >>= 1) {
      __syncthreads();
      const int i = tid, j = i ^ stride;
      if (j > i) {
        const bool desc = ((i & size) == 0);
        const float a = skey[i], c = skey[j];
        if (desc ? (a < c) : (a > c)) {
          skey[i] = c; skey[j] = a;
          const int t2 = sidx[i]; sidx[i] = sidx[j]; sidx[j] = t2;
        }
      }
    }
  }
  __syncthreads();

  const float sc = skey[tid];
  const int idx = sidx[tid];
  const float kx = (float)(idx & 511);
  const float ky = (float)(idx >> 9);
  const bool valid = sc > 0.0005f;
  const int o = b * KMAX + tid;
  kpts_out[o * 2 + 0] = valid ? kx : -1.0f;
  kpts_out[o * 2 + 1] = valid ? ky : -1.0f;
  scores_out[o] = valid ? sc : 0.0f;
  const float gx = ((kx - 3.5f) / 507.5f) * 2.0f - 1.0f;
  const float gy = ((ky - 3.5f) / 507.5f) * 2.0f - 1.0f;
  kinfo[o * 3 + 0] = (gx + 1.0f) * 0.5f * 63.0f;
  kinfo[o * 3 + 1] = (gy + 1.0f) * 0.5f * 63.0f;
  kinfo[o * 3 + 2] = valid ? 1.0f : 0.0f;
}

// ---------------------------------------------------------------------------
// Bilinear sample 256-d descriptor per keypoint (NHWC dmap) + L2 renormalize.
// ---------------------------------------------------------------------------
__global__ __launch_bounds__(256)
void sample_desc_kernel(const float* __restrict__ dmap, const float* __restrict__ kinfo,
                        float* __restrict__ desc_out)
{
  const int bk = blockIdx.x;
  const int b  = bk >> 10;
  const float ix = kinfo[bk * 3 + 0];
  const float iy = kinfo[bk * 3 + 1];
  const float vf = kinfo[bk * 3 + 2];
  const float x0f = floorf(ix), y0f = floorf(iy);
  const float wx1 = ix - x0f, wx0 = 1.0f - wx1;
  const float wy1 = iy - y0f, wy0 = 1.0f - wy1;
  const int ch = threadIdx.x;
  const float* dm = dmap + (size_t)b * 4096 * 256 + ch;

  auto g = [&](float yf, float xf) -> float {
    const bool ok = (xf >= 0.0f) && (xf <= 63.0f) && (yf >= 0.0f) && (yf <= 63.0f);
    const int xi = (int)fminf(fmaxf(xf, 0.0f), 63.0f);
    const int yi = (int)fminf(fmaxf(yf, 0.0f), 63.0f);
    return ok ? dm[(size_t)(yi * 64 + xi) * 256] : 0.0f;
  };
  const float v = g(y0f, x0f)             * wy0 * wx0
                + g(y0f, x0f + 1.f)       * wy0 * wx1
                + g(y0f + 1.f, x0f)       * wy1 * wx0
                + g(y0f + 1.f, x0f + 1.f) * wy1 * wx1;

  __shared__ float red[256];
  red[ch] = v * v;
  __syncthreads();
  for (int s = 128; s > 0; s >>= 1) {
    if (ch < s) red[ch] += red[ch + s];
    __syncthreads();
  }
  const float inv = 1.0f / fmaxf(sqrtf(red[0]), 1e-12f);
  desc_out[(size_t)bk * 256 + ch] = (vf > 0.5f) ? v * inv : 0.0f;
}

// ---------------------------------------------------------------------------
// Host orchestration
// ---------------------------------------------------------------------------
extern "C" void kernel_launch(void* const* d_in, const int* in_sizes, int n_in,
                              void* d_out, int out_size, void* d_ws, size_t ws_size,
                              hipStream_t stream)
{
  (void)in_sizes; (void)n_in; (void)out_size; (void)ws_size;

  const float* image = (const float*)d_in[0];
  const float* Wp[12]; const float* Bp[12];
  for (int i = 0; i < 12; ++i) { Wp[i] = (const float*)d_in[1 + 2 * i]; Bp[i] = (const float*)d_in[2 + 2 * i]; }
  // 0=w1a 1=w1b 2=w2a 3=w2b 4=w3a 5=w3b 6=w4a 7=w4b 8=wPa 9=wPb 10=wDa 11=wDb

  char* ws = (char*)d_ws;
  size_t off = 0;
  auto alloc = [&](size_t bytes) -> void* {
    void* p = ws + off;
    off = (off + bytes + 255) & ~(size_t)255;
    return p;
  };

  const int cin_[12]  = {1, 64, 64, 64, 64, 128, 128, 128, 128, 256, 128, 256};
  const int ks_[12]   = {3, 3, 3, 3, 3, 3, 3, 3, 3, 1, 3, 1};
  const int cout_[12] = {64, 64, 64, 64, 128, 128, 128, 128, 256, 65, 256, 256};

  _Float16* pk[12];
  for (int i = 0; i < 12; ++i) {
    const int kt = cin_[i] * ks_[i] * ks_[i];
    const int kp = (kt + 31) & ~31;
    const int coutPad = ((cout_[i] + MB - 1) / MB) * MB;
    pk[i] = (_Float16*)alloc((size_t)kp * coutPad * sizeof(_Float16));
  }

  _Float16* actA = (_Float16*)alloc((size_t)2 * 512 * 512 * 64 * sizeof(_Float16));
  _Float16* actB = (_Float16*)alloc((size_t)2 * 512 * 512 * 64 * sizeof(_Float16));
  float* dmap  = (float*)alloc((size_t)2 * 64 * 64 * 256 * sizeof(float));
  float* pmap  = (float*)alloc((size_t)2 * 64 * 64 * 65 * sizeof(float));
  float* smap  = (float*)alloc((size_t)2 * 512 * 512 * sizeof(float));
  float* kinfo = (float*)alloc((size_t)2 * KMAX * 3 * sizeof(float));

  for (int i = 0; i < 12; ++i) {
    const int kt = cin_[i] * ks_[i] * ks_[i];
    const int kp = (kt + 31) & ~31;
    const int coutPad = ((cout_[i] + MB - 1) / MB) * MB;
    const int total = kp * coutPad;
    pack_weights_kernel<<<(total + 255) / 256, 256, 0, stream>>>(
        Wp[i], pk[i], cout_[i], coutPad, cin_[i], ks_[i]);
  }

#define CONV(CIN, KS, INF32, RELU, OUTF32, IN, L, OUT, H, W)                         \
  conv_wmma_kernel<CIN, KS, INF32, RELU, OUTF32>                                     \
      <<<dim3((W) / TN, (H), 2 * ((cout_[L] + MB - 1) / MB)), 256, 0, stream>>>(     \
          (IN), pk[L], Bp[L], (OUT), (H), (W), cout_[L], (cout_[L] + MB - 1) / MB)

#define POOL(IN, OUT, C, HI)                                                         \
  maxpool_f16_kernel<<<((size_t)2 * (C) * ((HI) / 2) * ((HI) / 2) + 255) / 256, 256, \
                       0, stream>>>((IN), (OUT), (C), (HI), (HI))

  CONV(1,   3, true,  true,  false, image, 0, actA, 512, 512);   // conv1a
  CONV(64,  3, false, true,  false, actA,  1, actB, 512, 512);   // conv1b
  POOL(actB, actA, 64, 512);
  CONV(64,  3, false, true,  false, actA,  2, actB, 256, 256);   // conv2a
  CONV(64,  3, false, true,  false, actB,  3, actA, 256, 256);   // conv2b
  POOL(actA, actB, 64, 256);
  CONV(64,  3, false, true,  false, actB,  4, actA, 128, 128);   // conv3a
  CONV(128, 3, false, true,  false, actA,  5, actB, 128, 128);   // conv3b
  POOL(actB, actA, 128, 128);
  CONV(128, 3, false, true,  false, actA,  6, actB, 64, 64);     // conv4a
  CONV(128, 3, false, true,  false, actB,  7, actA, 64, 64);     // conv4b -> x in actA
  CONV(128, 3, false, true,  false, actA, 10, actB, 64, 64);     // wDa
  CONV(256, 1, false, false, true,  actB, 11, dmap, 64, 64);     // wDb -> d (f32 NHWC)
  CONV(128, 3, false, true,  false, actA,  8, actB, 64, 64);     // wPa
  CONV(256, 1, false, false, true,  actB,  9, pmap, 64, 64);     // wPb -> p (f32 NHWC)
#undef CONV
#undef POOL

  normalize_d_kernel<<<(2 * 64 * 64 + 255) / 256, 256, 0, stream>>>(dmap);
  softmax_d2s_kernel<<<(2 * 64 * 64 + 255) / 256, 256, 0, stream>>>(pmap, smap);

  float* out = (float*)d_out;
  float* out_kpts   = out;                           // [2][1024][2]
  float* out_scores = out + 2 * KMAX * 2;            // [2][1024]
  float* out_desc   = out + 2 * KMAX * 2 + 2 * KMAX; // [2][1024][256]

  topk_kernel<<<2, 1024, 0, stream>>>(smap, out_kpts, out_scores, kinfo);
  sample_desc_kernel<<<2 * KMAX, 256, 0, stream>>>(dmap, kinfo, out_desc);
}
